// LocalConvBlockv1_50010599194634
// MI455X (gfx1250) — matrix operand order repaired and checked by reading, not compiled, not run on hardware
//
#include <hip/hip_runtime.h>
#include <hip/hip_bf16.h>

// ---------------------------------------------------------------------------
// LocalConvBlock for MI455X (gfx1250): bf16 WMMA convs + bandwidth paste.
//   B=32, C=320, H=W=64, CTX_DIM=1024, CTX_HW=16
// 2x2 tile-blocked WMMA GEMM with zero-halo padded activations: the inner
// K-loop is pure {8x global_load_b128 -> 4x v_wmma_f32_16x16x32_bf16}.
// ---------------------------------------------------------------------------

typedef __bf16 bf16_t;
typedef __attribute__((ext_vector_type(16))) __bf16 v16bf;
typedef __attribute__((ext_vector_type(8)))  float  v8f;

#define B_      32
#define C_      320
#define H_      64
#define W_      64
#define CTXHW   16
#define PADHW   18           // 16 + 1-px zero halo on each side
#define CTXS    256          // 16*16 spatial
#define PADS    324          // 18*18
#define CTXC    1024         // context channels (indicator channels handled analytically)
#define CINFULL 1026

union FragBF { v16bf v; uint4 u[2]; };
union Acc8   { v8f  v; float f[8]; };

// ---------------------------------------------------------------------------
// Prep kernels
// ---------------------------------------------------------------------------
// ctx (B,256,1024) fp32 -> zero-halo padded bf16 [B][18][18][1024]
__global__ void k_pad_ctx(const float* __restrict__ src, bf16_t* __restrict__ dst) {
  int i = blockIdx.x * blockDim.x + threadIdx.x;
  const int n = B_ * PADS * CTXC;
  if (i >= n) return;
  const int c  = i & (CTXC - 1);
  const int p  = (i >> 10) % PADS;        // padded spatial index
  const int b  = i / (PADS * CTXC);
  const int py = p / PADHW, px = p % PADHW;
  float v = 0.f;
  if (py >= 1 && py <= CTXHW && px >= 1 && px <= CTXHW)
    v = src[((size_t)b * CTXS + (py - 1) * CTXHW + (px - 1)) * CTXC + c];
  dst[i] = (bf16_t)v;
}

// zero-fill padded act1 (halo must be zero; interior is overwritten by conv_in)
__global__ void k_zero_act1(bf16_t* __restrict__ dst, int n) {
  int i = blockIdx.x * blockDim.x + threadIdx.x;
  if (i < n) dst[i] = (bf16_t)0.f;
}

__global__ void k_pack_win(const float* __restrict__ w, bf16_t* __restrict__ dst) {
  // dst[(t*320+co)*1024+ci] = w_in[co][ci][t]   (w_in: (320,1026,3,3))
  int i = blockIdx.x * blockDim.x + threadIdx.x;
  const int n = 9 * C_ * CTXC;
  if (i >= n) return;
  int ci = i & (CTXC - 1);
  int co = (i >> 10) % C_;
  int t  = i / (C_ * CTXC);
  dst[i] = (bf16_t)w[((size_t)co * CINFULL + ci) * 9 + t];
}

__global__ void k_pack_wout(const float* __restrict__ w, bf16_t* __restrict__ dst) {
  // dst[(t*320+co)*320+ci] = w_out[co][ci][t]   (w_out: (320,320,3,3))
  int i = blockIdx.x * blockDim.x + threadIdx.x;
  const int n = 9 * C_ * C_;
  if (i >= n) return;
  int ci = i % C_;
  int co = (i / C_) % C_;
  int t  = i / (C_ * C_);
  dst[i] = (bf16_t)w[((size_t)co * C_ + ci) * 9 + t];
}

// ---------------------------------------------------------------------------
// Conv #1: (B,1026,16,16) --3x3--> (B,320,16,16), + bias + indicator + SiLU,
// output bf16 into zero-halo padded act1 [B][18][18][320].
// One wave per (batch, 2-row tile, 32-channel tile): 2x2 WMMA blocking.
// ---------------------------------------------------------------------------
__global__ void __launch_bounds__(32)
k_conv_in(const bf16_t* __restrict__ ctxb,      // [B][18][18][1024] bf16 (halo=0)
          const bf16_t* __restrict__ wpin,      // [9][320][1024] bf16
          const float*  __restrict__ b_in,      // [320]
          const float*  __restrict__ w_raw,     // original w_in (320,1026,3,3) fp32
          const float*  __restrict__ indicator, // [B][2]
          bf16_t* __restrict__ act1)            // [B][18][18][320] bf16 (halo=0)
{
  const int nt   = blockIdx.x;         // 0..9  -> channels nt*32 + {0..15, 16..31}
  const int mb   = blockIdx.y;         // 0..7  -> spatial rows {2mb, 2mb+1}
  const int bz   = blockIdx.z;         // batch
  const int lane = threadIdx.x;
  const int half = lane >> 4;
  const int nn   = lane & 15;
  const int co0  = nt * 32 + nn;
  const int co1  = co0 + 16;
  const int y0   = mb * 2;

  Acc8 acc[2][2];                      // [row-tile][chan-tile]
#pragma unroll
  for (int mi = 0; mi < 2; ++mi)
#pragma unroll
    for (int ni = 0; ni < 2; ++ni)
      acc[mi][ni].v = (v8f){0.f,0.f,0.f,0.f,0.f,0.f,0.f,0.f};

  for (int t = 0; t < 9; ++t) {
    const int dy = t / 3, dx = t % 3;
    // padded coords: halo offset +1 already folded in (iy = y0+dy-1+1)
    const int py0 = y0 + dy;
    const int px  = nn + dx;
    const bf16_t* arow0 = ctxb + ((size_t)((bz * PADHW + py0) * PADHW + px) * CTXC);
    const bf16_t* arow1 = arow0 + (size_t)PADHW * CTXC;   // row py0+1
    const bf16_t* wr0   = wpin + ((size_t)(t * C_ + co0) * CTXC) + half * 16;
    const bf16_t* wr1   = wpin + ((size_t)(t * C_ + co1) * CTXC) + half * 16;

#pragma unroll 2
    for (int ch = 0; ch < 32; ++ch) {
      const int c0 = ch * 32;
      const uint4* ap0 = (const uint4*)(arow0 + c0);
      const uint4* ap1 = (const uint4*)(arow1 + c0);
      const uint4* bp0 = (const uint4*)(wr0 + c0);
      const uint4* bp1 = (const uint4*)(wr1 + c0);
      FragBF a0, a1, b0, b1;
      a0.u[0] = ap0[2 * half];  a0.u[1] = ap0[2 * half + 1];
      a1.u[0] = ap1[2 * half];  a1.u[1] = ap1[2 * half + 1];
      b0.u[0] = bp0[0];     b0.u[1] = bp0[1];
      b1.u[0] = bp1[0];     b1.u[1] = bp1[1];
      acc[0][0].v = __builtin_amdgcn_wmma_f32_16x16x32_bf16(
          false, a0.v, false, b0.v, (short)0, acc[0][0].v, false, false);
      acc[0][1].v = __builtin_amdgcn_wmma_f32_16x16x32_bf16(
          false, a0.v, false, b1.v, (short)0, acc[0][1].v, false, false);
      acc[1][0].v = __builtin_amdgcn_wmma_f32_16x16x32_bf16(
          false, a1.v, false, b0.v, (short)0, acc[1][0].v, false, false);
      acc[1][1].v = __builtin_amdgcn_wmma_f32_16x16x32_bf16(
          false, a1.v, false, b1.v, (short)0, acc[1][1].v, false, false);
    }
  }

  // Epilogue: bias + analytic indicator-channel contribution + SiLU -> bf16
  const float i0 = indicator[bz * 2 + 0];
  const float i1 = indicator[bz * 2 + 1];
  float tw[2][9];
#pragma unroll
  for (int ni = 0; ni < 2; ++ni) {
    const int co = ni ? co1 : co0;
#pragma unroll
    for (int t = 0; t < 9; ++t)
      tw[ni][t] = i0 * w_raw[((size_t)co * CINFULL + 1024) * 9 + t]
                + i1 * w_raw[((size_t)co * CINFULL + 1025) * 9 + t];
  }
  const float bia[2] = { b_in[co0], b_in[co1] };

#pragma unroll
  for (int mi = 0; mi < 2; ++mi) {
    const int y = y0 + mi;
#pragma unroll
    for (int ni = 0; ni < 2; ++ni) {
      const int co = ni ? co1 : co0;
#pragma unroll
      for (int i = 0; i < 8; ++i) {
        const int xo = i + half * 8;               // D-matrix row M -> spatial x
        float extra = 0.f;
#pragma unroll
        for (int t = 0; t < 9; ++t) {
          const int iy = y + t / 3 - 1, ix = xo + t % 3 - 1;
          if (iy >= 0 && iy < CTXHW && ix >= 0 && ix < CTXHW) extra += tw[ni][t];
        }
        const float v = acc[mi][ni].f[i] + bia[ni] + extra;
        const float s = v / (1.f + __expf(-v));    // SiLU
        act1[((size_t)((bz * PADHW + y + 1) * PADHW) + xo + 1) * C_ + co] = (bf16_t)s;
      }
    }
  }
}

// ---------------------------------------------------------------------------
// Conv #2: (B,320,16,16) --3x3--> (B,320,16,16) + bias, fp32 out cx2[b][s][co]
// ---------------------------------------------------------------------------
__global__ void __launch_bounds__(32)
k_conv_out(const bf16_t* __restrict__ act1,   // [B][18][18][320] bf16 (halo=0)
           const bf16_t* __restrict__ wpout,  // [9][320][320] bf16
           const float*  __restrict__ b_out,  // [320]
           float* __restrict__ cx2)           // [B][256][320] fp32
{
  const int nt   = blockIdx.x;         // 0..9
  const int mb   = blockIdx.y;         // 0..7
  const int bz   = blockIdx.z;
  const int lane = threadIdx.x;
  const int half = lane >> 4;
  const int nn   = lane & 15;
  const int co0  = nt * 32 + nn;
  const int co1  = co0 + 16;
  const int y0   = mb * 2;

  Acc8 acc[2][2];
#pragma unroll
  for (int mi = 0; mi < 2; ++mi)
#pragma unroll
    for (int ni = 0; ni < 2; ++ni)
      acc[mi][ni].v = (v8f){0.f,0.f,0.f,0.f,0.f,0.f,0.f,0.f};

  for (int t = 0; t < 9; ++t) {
    const int dy = t / 3, dx = t % 3;
    const int py0 = y0 + dy;
    const int px  = nn + dx;
    const bf16_t* arow0 = act1 + ((size_t)((bz * PADHW + py0) * PADHW + px) * C_);
    const bf16_t* arow1 = arow0 + (size_t)PADHW * C_;
    const bf16_t* wr0   = wpout + ((size_t)(t * C_ + co0) * C_) + half * 16;
    const bf16_t* wr1   = wpout + ((size_t)(t * C_ + co1) * C_) + half * 16;

#pragma unroll 2
    for (int ch = 0; ch < 10; ++ch) {             // K = 320 per tap
      const int c0 = ch * 32;
      const uint4* ap0 = (const uint4*)(arow0 + c0);
      const uint4* ap1 = (const uint4*)(arow1 + c0);
      const uint4* bp0 = (const uint4*)(wr0 + c0);
      const uint4* bp1 = (const uint4*)(wr1 + c0);
      FragBF a0, a1, b0, b1;
      a0.u[0] = ap0[2 * half];  a0.u[1] = ap0[2 * half + 1];
      a1.u[0] = ap1[2 * half];  a1.u[1] = ap1[2 * half + 1];
      b0.u[0] = bp0[0];     b0.u[1] = bp0[1];
      b1.u[0] = bp1[0];     b1.u[1] = bp1[1];
      acc[0][0].v = __builtin_amdgcn_wmma_f32_16x16x32_bf16(
          false, a0.v, false, b0.v, (short)0, acc[0][0].v, false, false);
      acc[0][1].v = __builtin_amdgcn_wmma_f32_16x16x32_bf16(
          false, a0.v, false, b1.v, (short)0, acc[0][1].v, false, false);
      acc[1][0].v = __builtin_amdgcn_wmma_f32_16x16x32_bf16(
          false, a1.v, false, b0.v, (short)0, acc[1][0].v, false, false);
      acc[1][1].v = __builtin_amdgcn_wmma_f32_16x16x32_bf16(
          false, a1.v, false, b1.v, (short)0, acc[1][1].v, false, false);
    }
  }

  const float bia[2] = { b_out[co0], b_out[co1] };
#pragma unroll
  for (int mi = 0; mi < 2; ++mi) {
    const int y = y0 + mi;
#pragma unroll
    for (int ni = 0; ni < 2; ++ni) {
      const int co = ni ? co1 : co0;
#pragma unroll
      for (int i = 0; i < 8; ++i) {
        const int xo = i + half * 8;
        cx2[((size_t)(bz * CTXS) + y * CTXHW + xo) * C_ + co] = acc[mi][ni].f[i] + bia[ni];
      }
    }
  }
}

// ---------------------------------------------------------------------------
// Paste: out = global_x + mask * cx2[b, :, src_y, src_x]  (bandwidth bound)
// One thread per (b, c, y, 4 x's) with float4 traffic.
// ---------------------------------------------------------------------------
__global__ void k_paste(const float* __restrict__ gx,
                        const float* __restrict__ cx2,
                        const float* __restrict__ bbox,
                        float* __restrict__ out)
{
  const int idx = blockIdx.x * blockDim.x + threadIdx.x;
  const int total = B_ * C_ * H_ * (W_ / 4);
  if (idx >= total) return;
  const int x4 = idx & 15;
  const int y  = (idx >> 4) & 63;
  const int c  = (idx >> 10) % C_;
  const int b  = idx / (16 * 64 * C_);

  const float4 bb = ((const float4*)bbox)[b];
  const int x1 = (int)(bb.x * (float)H_);
  const int y1 = (int)(bb.y * (float)H_);
  const int x2 = max((int)(bb.z * (float)H_), x1 + 1);
  const int y2 = max((int)(bb.w * (float)H_), y1 + 1);
  const bool ok = (y2 <= H_) && (x2 <= W_) && (y1 >= 0) && (x1 >= 0);
  const bool vy = (y >= y1) && (y < y2);

  float4 r = ((const float4*)gx)[idx];
  if (ok && vy) {
    int sy = (y - y1) * CTXHW / (y2 - y1);
    sy = min(max(sy, 0), CTXHW - 1);
    const float* crow = cx2 + ((size_t)b * CTXS + sy * CTXHW) * C_ + c;
    const int xb = x4 * 4;
#pragma unroll
    for (int j = 0; j < 4; ++j) {
      const int xx = xb + j;
      if (xx >= x1 && xx < x2) {
        int sx = (xx - x1) * CTXHW / (x2 - x1);
        sx = min(max(sx, 0), CTXHW - 1);
        ((float*)&r)[j] += crow[sx * C_];
      }
    }
  }
  ((float4*)out)[idx] = r;
}

// ---------------------------------------------------------------------------
extern "C" void kernel_launch(void* const* d_in, const int* in_sizes, int n_in,
                              void* d_out, int out_size, void* d_ws, size_t ws_size,
                              hipStream_t stream) {
  const float* gx    = (const float*)d_in[0];
  const float* ctx   = (const float*)d_in[1];
  const float* ind   = (const float*)d_in[2];
  const float* bbox  = (const float*)d_in[3];
  const float* w_in  = (const float*)d_in[4];
  const float* b_in  = (const float*)d_in[5];
  const float* w_out = (const float*)d_in[6];
  const float* b_out = (const float*)d_in[7];
  float* out = (float*)d_out;

  char*  ws  = (char*)d_ws;
  size_t off = 0;
  auto alloc = [&](size_t bytes) -> void* {
    void* p = ws + off;
    off += (bytes + 255) & ~(size_t)255;
    return p;
  };
  bf16_t* ctxb  = (bf16_t*)alloc((size_t)B_ * PADS * CTXC * sizeof(bf16_t)); // 21.2 MB
  bf16_t* wpin  = (bf16_t*)alloc((size_t)9 * C_ * CTXC * sizeof(bf16_t));    //  5.9 MB
  bf16_t* wpout = (bf16_t*)alloc((size_t)9 * C_ * C_   * sizeof(bf16_t));    //  1.8 MB
  bf16_t* act1  = (bf16_t*)alloc((size_t)B_ * PADS * C_ * sizeof(bf16_t));   //  6.6 MB
  float*  cx2   = (float*) alloc((size_t)B_ * CTXS * C_ * sizeof(float));    // 10.5 MB

  // --- prep ---
  {
    const int n = B_ * PADS * CTXC;
    k_pad_ctx<<<(n + 255) / 256, 256, 0, stream>>>(ctx, ctxb);
  }
  {
    const int n = B_ * PADS * C_;
    k_zero_act1<<<(n + 255) / 256, 256, 0, stream>>>(act1, n);
  }
  {
    const int n = 9 * C_ * CTXC;
    k_pack_win<<<(n + 255) / 256, 256, 0, stream>>>(w_in, wpin);
  }
  {
    const int n = 9 * C_ * C_;
    k_pack_wout<<<(n + 255) / 256, 256, 0, stream>>>(w_out, wpout);
  }

  // --- conv_in + SiLU (bf16 WMMA, 2x2 blocked) ---
  {
    dim3 grid(C_ / 32, CTXHW / 2, B_);   // (10, 8, 32)
    k_conv_in<<<grid, 32, 0, stream>>>(ctxb, wpin, b_in, w_in, ind, act1);
  }

  // --- conv_out (bf16 WMMA, 2x2 blocked) ---
  {
    dim3 grid(C_ / 32, CTXHW / 2, B_);
    k_conv_out<<<grid, 32, 0, stream>>>(act1, wpout, b_out, cx2);
  }

  // --- bbox paste (bandwidth) ---
  {
    const int total = B_ * C_ * H_ * (W_ / 4);
    k_paste<<<(total + 255) / 256, 256, 0, stream>>>(gx, cx2, bbox, out);
  }
}
